// LilMultiheadAttention_14886356648754
// MI455X (gfx1250) — matrix operand order repaired and checked by reading, not compiled, not run on hardware
//
#include <hip/hip_runtime.h>

// ---------------------------------------------------------------------------
// MI455X (gfx1250) causal multi-head attention, bf16 WMMA path.
//   D=768, H=12, hw=64, B=2, S=2048
// Pipeline:
//   0) f32 -> bf16 convert of X and all weights (one conversion per element)
//   1) fused QKV projection (16x64 tile per wave, software-pipelined k-loop),
//      V stored transposed [B,H,64,S] for contiguous PV fragment loads
//   2) flash attention, one wave per (b,h,16-row q tile); K/V fragment loads
//      hoisted so they overlap the softmax VALU work
//   3) output projection + bias (16x64 tile per wave, software-pipelined)
// ---------------------------------------------------------------------------

typedef __attribute__((ext_vector_type(16))) __bf16         v16bf;
typedef __attribute__((ext_vector_type(8)))  float          v8f;
typedef __attribute__((ext_vector_type(4)))  float          f32x4;
typedef __attribute__((ext_vector_type(4)))  unsigned short u16x4;
typedef __attribute__((ext_vector_type(8)))  unsigned short u16x8;
typedef __attribute__((ext_vector_type(16))) unsigned short u16x16;

#define D_MODEL 768
#define SEQ     2048
#define NHEAD   12
#define HDIM    64

__device__ __forceinline__ unsigned short f2bf(float f) {
  return __builtin_bit_cast(unsigned short, (__bf16)f);   // native cvt
}
__device__ __forceinline__ v16bf as_bf(u16x16 u) {
  return __builtin_bit_cast(v16bf, u);
}
__device__ __forceinline__ u16x16 join16(u16x8 lo, u16x8 hi) {
  u16x16 r;
#pragma unroll
  for (int i = 0; i < 8; ++i) { r[i] = lo[i]; r[i + 8] = hi[i]; }
  return r;
}
// A-fragment: two 8-element runs 16 elements apart (k = 16*(h>=8)+8*hi+(h&7))
__device__ __forceinline__ v16bf loadA(const unsigned short* p) {
  return as_bf(join16(*(const u16x8*)(p), *(const u16x8*)(p + 16)));
}
// B-fragment: two 8-element runs 16 elements apart (k = 16*(h>=8)+8*hi+(h&7))
__device__ __forceinline__ v16bf loadB(const unsigned short* p) {
  return as_bf(join16(*(const u16x8*)(p), *(const u16x8*)(p + 16)));
}
__device__ __forceinline__ v8f mma(v16bf a, v16bf b, v8f c) {
  return __builtin_amdgcn_wmma_f32_16x16x32_bf16(false, a, false, b,
                                                 (short)0, c, false, false);
}

// ---------------------------------------------------------------------------
// Kernel 0: elementwise f32 -> bf16 (4 elements / thread)
// ---------------------------------------------------------------------------
__global__ void cvt_bf16_kernel(const float* __restrict__ in,
                                unsigned short* __restrict__ out, int n4) {
  const int i = blockIdx.x * blockDim.x + threadIdx.x;
  if (i >= n4) return;
  f32x4 v = ((const f32x4*)in)[i];
  u16x4 o;
#pragma unroll
  for (int j = 0; j < 4; ++j) o[j] = f2bf(v[j]);
  ((u16x4*)out)[i] = o;
}

// ---------------------------------------------------------------------------
// Kernel 1: fused QKV projection from bf16 inputs. Y = X @ W^T.
// One wave: 16(M) x 64(N) tile; k-loop double-buffered (loads one step ahead).
// Q pre-scaled by (1/8)*log2(e).  Q,K -> [B,H,S,64]; V -> [B,H,64,S] (T).
// ---------------------------------------------------------------------------
__global__ void qkv_proj_kernel(const unsigned short* __restrict__ Xb,
                                const unsigned short* __restrict__ Wqb,
                                const unsigned short* __restrict__ Wkb,
                                const unsigned short* __restrict__ Wvb,
                                unsigned short* __restrict__ Qb,
                                unsigned short* __restrict__ Kb,
                                unsigned short* __restrict__ Vt) {
  const int lane = threadIdx.x & 31;
  const int lr   = lane & 15;
  const int hi   = (lane & 16) ? 1 : 0;

  const int m0  = blockIdx.x * 16;              // row tile   [0,4096)
  const int mat = blockIdx.y / (D_MODEL / 64);  // 0:Q 1:K 2:V
  const int n0  = (blockIdx.y % (D_MODEL / 64)) * 64;

  const unsigned short* W = (mat == 0) ? Wqb : (mat == 1) ? Wkb : Wvb;
  const unsigned short* ap = Xb + (m0 + lr) * D_MODEL + hi * 8;
  const unsigned short* bp = W + (n0 + lr) * D_MODEL + hi * 8;

  v8f acc[4] = {};
  v16bf af = loadA(ap);
  v16bf bf[4];
#pragma unroll
  for (int c = 0; c < 4; ++c) bf[c] = loadB(bp + c * 16 * D_MODEL);

  for (int k0 = 0; k0 < D_MODEL - 32; k0 += 32) {
    // issue next k-step's loads before consuming current fragments
    const v16bf afn = loadA(ap + k0 + 32);
    v16bf bfn[4];
#pragma unroll
    for (int c = 0; c < 4; ++c) bfn[c] = loadB(bp + c * 16 * D_MODEL + k0 + 32);
#pragma unroll
    for (int c = 0; c < 4; ++c) acc[c] = mma(af, bf[c], acc[c]);
    af = afn;
#pragma unroll
    for (int c = 0; c < 4; ++c) bf[c] = bfn[c];
  }
#pragma unroll
  for (int c = 0; c < 4; ++c) acc[c] = mma(af, bf[c], acc[c]);

  const int hd = n0 >> 6;                       // head (tile is 64-aligned)
  const float qs = (mat == 0) ? 0.18033688f : 1.0f;  // (1/8)*log2(e)
#pragma unroll
  for (int c = 0; c < 4; ++c) {
    const int dcol = c * 16 + lr;               // dim within head
#pragma unroll
    for (int r = 0; r < 8; ++r) {
      const int row = m0 + r + hi * 8;          // global token index
      const int bb  = row >> 11;
      const int ss  = row & 2047;
      const int bh  = bb * NHEAD + hd;
      const unsigned short val = f2bf(acc[c][r] * qs);
      if (mat == 0)      Qb[((size_t)bh * SEQ + ss) * HDIM + dcol] = val;
      else if (mat == 1) Kb[((size_t)bh * SEQ + ss) * HDIM + dcol] = val;
      else               Vt[((size_t)bh * HDIM + dcol) * SEQ + ss] = val; // T
    }
  }
}

// ---------------------------------------------------------------------------
// Kernel 2: causal flash attention.  One wave per (b, h, 16-row q tile).
// kv in blocks of 32; online softmax in base-2 (scale folded into Q).
// All K and V fragments for the block are loaded up front so the V loads
// stay in flight across the softmax VALU section and LDS relayout.
// ---------------------------------------------------------------------------
#define ATT_WAVES 8
__global__ void attn_kernel(const unsigned short* __restrict__ Qb,
                            const unsigned short* __restrict__ Kb,
                            const unsigned short* __restrict__ Vt,
                            unsigned short* __restrict__ Ctx) {
  __shared__ unsigned short smem[ATT_WAVES][16][40];   // pitch 80B (16B align)

  const int lane  = threadIdx.x & 31;
  const int wslot = threadIdx.x >> 5;
  const int lr    = lane & 15;
  const int hi    = (lane & 16) ? 1 : 0;

  const int wid = blockIdx.x * ATT_WAVES + wslot;
  const int b   = wid / (NHEAD * (SEQ / 16));
  const int hq  = wid % (NHEAD * (SEQ / 16));
  const int h   = hq / (SEQ / 16);
  const int q0  = (hq % (SEQ / 16)) * 16;

  const unsigned short* Qh = Qb + (size_t)(b * NHEAD + h) * SEQ * HDIM;
  const unsigned short* Kh = Kb + (size_t)(b * NHEAD + h) * SEQ * HDIM;
  const unsigned short* Vh = Vt + (size_t)(b * NHEAD + h) * HDIM * SEQ;

  // Q fragments for the whole kv loop (16x64 = two 16x32 A fragments)
  v16bf qf[2];
#pragma unroll
  for (int c = 0; c < 2; ++c) qf[c] = loadA(Qh + (q0 + lr) * HDIM + c * 32 + hi * 8);

  v8f acc[4] = {};                 // ctx accumulator 16x64
  float mrow[8], lsum[8];
#pragma unroll
  for (int r = 0; r < 8; ++r) { mrow[r] = -3.0e38f; lsum[r] = 0.0f; }

  for (int kv0 = 0; kv0 <= q0 + 15; kv0 += 32) {
    // ---- issue all K and V fragment loads for this block ----
    v16bf kf[4], vf[4];
#pragma unroll
    for (int c = 0; c < 2; ++c) {
      const unsigned short* kp = Kh + (kv0 + lr) * HDIM + c * 32 + hi * 8;
      kf[c * 2 + 0] = loadB(kp);
      kf[c * 2 + 1] = loadB(kp + 16 * HDIM);
    }
#pragma unroll
    for (int c = 0; c < 4; ++c)
      vf[c] = loadB(Vh + (size_t)(c * 16 + lr) * SEQ + kv0 + hi * 8);

    // ---- scores: two 16x16 tiles of Q @ K^T (K-dim 64 = 2 chunks) ----
    v8f s0 = {}, s1 = {};
    s0 = mma(qf[0], kf[0], s0);
    s1 = mma(qf[0], kf[1], s1);
    s0 = mma(qf[1], kf[2], s0);
    s1 = mma(qf[1], kf[3], s1);

    // ---- causal mask + online softmax (row stats across 16-lane groups) ----
    float p0[8], p1[8];
#pragma unroll
    for (int r = 0; r < 8; ++r) {
      const int qidx = q0 + r + hi * 8;
      const int k0i  = kv0 + lr;
      float v0 = (k0i <= qidx)      ? s0[r] : -3.0e38f;
      float v1 = (k0i + 16 <= qidx) ? s1[r] : -3.0e38f;
      float rm = fmaxf(v0, v1);
      rm = fmaxf(rm, __shfl_xor(rm, 1));
      rm = fmaxf(rm, __shfl_xor(rm, 2));
      rm = fmaxf(rm, __shfl_xor(rm, 4));
      rm = fmaxf(rm, __shfl_xor(rm, 8));
      const float mnew = fmaxf(mrow[r], rm);
      const float a    = exp2f(mrow[r] - mnew);
      p0[r] = exp2f(v0 - mnew);
      p1[r] = exp2f(v1 - mnew);
      float ps = p0[r] + p1[r];
      ps += __shfl_xor(ps, 1);
      ps += __shfl_xor(ps, 2);
      ps += __shfl_xor(ps, 4);
      ps += __shfl_xor(ps, 8);
      lsum[r] = lsum[r] * a + ps;
      mrow[r] = mnew;
      acc[0][r] *= a; acc[1][r] *= a; acc[2][r] *= a; acc[3][r] *= a;
    }

    // ---- P: C-layout -> LDS -> A-fragment layout ----
    unsigned short (*sm)[40] = smem[wslot];
#pragma unroll
    for (int r = 0; r < 8; ++r) {
      const int row = r + hi * 8;
      sm[row][lr]      = f2bf(p0[r]);
      sm[row][16 + lr] = f2bf(p1[r]);
    }
    const unsigned short* pp = &sm[lr][hi * 8];
    const v16bf pf = as_bf(join16(*(const u16x8*)(pp), *(const u16x8*)(pp + 16)));

    // ---- ctx += P(16x32) @ V(32x64) ----
#pragma unroll
    for (int c = 0; c < 4; ++c) acc[c] = mma(pf, vf[c], acc[c]);
  }

  // ---- normalize and write ctx (recombined heads) as bf16 [B,S,768] ----
#pragma unroll
  for (int r = 0; r < 8; ++r) {
    const float inv = 1.0f / lsum[r];
    const int row = q0 + r + hi * 8;
    unsigned short* op = Ctx + (size_t)((b << 11) + row) * D_MODEL + h * HDIM;
#pragma unroll
    for (int c = 0; c < 4; ++c) op[c * 16 + lr] = f2bf(acc[c][r] * inv);
  }
}

// ---------------------------------------------------------------------------
// Kernel 3: output projection.  Out = ctx @ Wo^T + bo (f32 output).
// One wave: 16(M) x 64(N) tile; k-loop double-buffered.
// ---------------------------------------------------------------------------
__global__ void out_proj_kernel(const unsigned short* __restrict__ Ctx,
                                const unsigned short* __restrict__ Wob,
                                const float* __restrict__ bo,
                                float* __restrict__ Out) {
  const int lane = threadIdx.x & 31;
  const int lr   = lane & 15;
  const int hi   = (lane & 16) ? 1 : 0;
  const int m0 = blockIdx.x * 16;
  const int n0 = blockIdx.y * 64;

  const unsigned short* ap = Ctx + (m0 + lr) * D_MODEL + hi * 8;
  const unsigned short* bp = Wob + (n0 + lr) * D_MODEL + hi * 8;

  v8f acc[4] = {};
  v16bf af = loadA(ap);
  v16bf bf[4];
#pragma unroll
  for (int c = 0; c < 4; ++c) bf[c] = loadB(bp + c * 16 * D_MODEL);

  for (int k0 = 0; k0 < D_MODEL - 32; k0 += 32) {
    const v16bf afn = loadA(ap + k0 + 32);
    v16bf bfn[4];
#pragma unroll
    for (int c = 0; c < 4; ++c) bfn[c] = loadB(bp + c * 16 * D_MODEL + k0 + 32);
#pragma unroll
    for (int c = 0; c < 4; ++c) acc[c] = mma(af, bf[c], acc[c]);
    af = afn;
#pragma unroll
    for (int c = 0; c < 4; ++c) bf[c] = bfn[c];
  }
#pragma unroll
  for (int c = 0; c < 4; ++c) acc[c] = mma(af, bf[c], acc[c]);

#pragma unroll
  for (int c = 0; c < 4; ++c) {
    const int col = n0 + c * 16 + lr;
    const float bias = bo[col];
#pragma unroll
    for (int r = 0; r < 8; ++r) {
      Out[(size_t)(m0 + r + hi * 8) * D_MODEL + col] = acc[c][r] + bias;
    }
  }
}

// ---------------------------------------------------------------------------
extern "C" void kernel_launch(void* const* d_in, const int* in_sizes, int n_in,
                              void* d_out, int out_size, void* d_ws, size_t ws_size,
                              hipStream_t stream) {
  const float* X  = (const float*)d_in[0];
  const float* Wq = (const float*)d_in[1];
  const float* Wk = (const float*)d_in[2];
  const float* Wv = (const float*)d_in[3];
  const float* Wo = (const float*)d_in[4];
  const float* bo = (const float*)d_in[5];
  float* Out = (float*)d_out;

  // -------- workspace layout (bf16 stored as u16) --------
  const size_t NTOK   = (size_t)2 * SEQ;                 // 4096 tokens
  const size_t HELEMS = NTOK * NHEAD * HDIM;             // per Q/K/V: 3.1M
  const size_t XELEMS = NTOK * D_MODEL;                  // 3.1M
  const size_t WELEMS = (size_t)D_MODEL * D_MODEL;       // 589k
  char* ws = (char*)d_ws;
  unsigned short* Qb  = (unsigned short*)(ws);
  unsigned short* Kb  = Qb  + HELEMS;
  unsigned short* Vt  = Kb  + HELEMS;
  unsigned short* Cx  = Vt  + HELEMS;
  unsigned short* Xb  = Cx  + XELEMS;
  unsigned short* Wqb = Xb  + XELEMS;
  unsigned short* Wkb = Wqb + WELEMS;
  unsigned short* Wvb = Wkb + WELEMS;
  unsigned short* Wob = Wvb + WELEMS;

  // 0) convert inputs to bf16 (once per element)
  {
    const int xq = (int)(XELEMS / 4), wq = (int)(WELEMS / 4);
    cvt_bf16_kernel<<<(xq + 255) / 256, 256, 0, stream>>>(X,  Xb,  xq);
    cvt_bf16_kernel<<<(wq + 255) / 256, 256, 0, stream>>>(Wq, Wqb, wq);
    cvt_bf16_kernel<<<(wq + 255) / 256, 256, 0, stream>>>(Wk, Wkb, wq);
    cvt_bf16_kernel<<<(wq + 255) / 256, 256, 0, stream>>>(Wv, Wvb, wq);
    cvt_bf16_kernel<<<(wq + 255) / 256, 256, 0, stream>>>(Wo, Wob, wq);
  }

  // 1) fused QKV projection: 4096 x (3*768), 16x64 tile per wave
  qkv_proj_kernel<<<dim3(4096 / 16, 3 * (D_MODEL / 64)), 32, 0, stream>>>(
      Xb, Wqb, Wkb, Wvb, Qb, Kb, Vt);

  // 2) causal flash attention: one wave per (b,h,q-tile) = 3072 waves
  attn_kernel<<<dim3((2 * NHEAD * (SEQ / 16)) / ATT_WAVES), 32 * ATT_WAVES, 0,
                stream>>>(Qb, Kb, Vt, Cx);

  // 3) output projection + bias
  out_proj_kernel<<<dim3(4096 / 16, D_MODEL / 64), 32, 0, stream>>>(
      Cx, Wob, bo, Out);
}
